// LocalTransformerDecoder_12756052869792
// MI455X (gfx1250) — matrix operand order repaired and checked by reading, not compiled, not run on hardware
//
#include <hip/hip_runtime.h>
#include <hip/hip_bf16.h>
#include <math.h>

// ---------------- problem constants (match reference) ----------------
constexpr int CB  = 2;      // batch
constexpr int CS  = 2000;   // seq len
constexpr int CD  = 512;    // model dim
constexpr int CH  = 8;      // heads
constexpr int CHD = 64;     // head dim
constexpr int CMLP= 2048;
constexpr int CL  = 6;
constexpr int CV  = 32000;
constexpr int CBS = 50;     // local block size
constexpr int CNB = CS / CBS;        // 40
constexpr int CM  = CB * CS;         // 4000 token rows

typedef __bf16 bf16;
typedef __attribute__((ext_vector_type(16))) __bf16 v16bf;
typedef __attribute__((ext_vector_type(8)))  float  v8f;

// CDNA5 async global->LDS staging via inline asm (bypasses builtin pointer-AS
// typing). Device-pass only; host pass / other targets use the uint4 fallback.
#if defined(__AMDGCN__) && defined(__gfx1250__)
#define USE_ASYNC_LDS 1
#endif

// copy 16 bf16 (32 bytes) global -> LDS
__device__ __forceinline__ void copy16_g2l(bf16* dst_lds, const bf16* src_glob) {
#ifdef USE_ASYNC_LDS
  unsigned ldsa = (unsigned)(size_t)dst_lds;              // LDS byte address = addr[31:0]
  unsigned long long ga = (unsigned long long)(size_t)src_glob;
  // INST_OFFSET is added to both the LDS and global addresses (ISA 15.18.3),
  // so one address pair covers both 16B beats. Tracked by ASYNCcnt.
  asm volatile(
      "global_load_async_to_lds_b128 %0, %1, off\n\t"
      "global_load_async_to_lds_b128 %0, %1, off offset:16"
      :: "v"(ldsa), "v"(ga) : "memory");
#else
  uint4 a = *(const uint4*)src_glob;
  uint4 b = *(const uint4*)(src_glob + 8);
  ((uint4*)dst_lds)[0] = a;
  ((uint4*)dst_lds)[1] = b;
#endif
}

__device__ __forceinline__ void wait_stage() {
#ifdef USE_ASYNC_LDS
  asm volatile("s_wait_asynccnt 0x0" ::: "memory");
#endif
}

// ---------------- helpers ----------------
__device__ __forceinline__ float gelu_tanh(float x) {
  float x3 = x * x * x;
  return 0.5f * x * (1.0f + tanhf(0.7978845608028654f * (x + 0.044715f * x3)));
}

// ---------------- f32 -> bf16 cast (grid-stride) ----------------
__global__ void cast_f32_bf16(const float* __restrict__ in, bf16* __restrict__ out, size_t n) {
  size_t i = (size_t)blockIdx.x * blockDim.x + threadIdx.x;
  size_t stride = (size_t)gridDim.x * blockDim.x;
  for (; i < n; i += stride) out[i] = (bf16)in[i];
}

// ---------------- f32 [L][K][N] -> bf16 transposed [L][N][K] ----------------
__global__ __launch_bounds__(256) void cast_transpose_bf16(const float* __restrict__ in,
                                                           bf16* __restrict__ out,
                                                           int K, int N) {
  __shared__ float tile[32][33];
  int l = blockIdx.z;
  const float* src = in + (size_t)l * K * N;
  bf16*        dst = out + (size_t)l * K * N;
  int k0 = blockIdx.y * 32, n0 = blockIdx.x * 32;
  int tx = threadIdx.x & 31, ty = threadIdx.x >> 5;  // 8 rows per pass
#pragma unroll
  for (int r = ty; r < 32; r += 8)
    tile[r][tx] = src[(size_t)(k0 + r) * N + n0 + tx];
  __syncthreads();
#pragma unroll
  for (int r = ty; r < 32; r += 8)
    dst[(size_t)(n0 + r) * K + k0 + tx] = (bf16)tile[tx][r];
}

// ---------------- embedding + sinusoidal positional encoding ----------------
__global__ __launch_bounds__(256) void embed_pe(const int* __restrict__ inp,
                                                const float* __restrict__ emb,
                                                float* __restrict__ x) {
  int row = blockIdx.x;              // 0..CM-1 ; row = b*CS + s
  int s   = row % CS;
  int tok = inp[row];
  float pos = (float)s;
  const float c = -logf(10000.0f) / (float)CD;
  for (int j = threadIdx.x; j < CD / 2; j += blockDim.x) {
    float div = expf(2.0f * (float)j * c);
    float a = pos * div;
    size_t o = (size_t)row * CD + 2 * j;
    size_t e = (size_t)tok * CD + 2 * j;
    x[o]     = emb[e]     + sinf(a);
    x[o + 1] = emb[e + 1] + cosf(a);
  }
}

// ---------------- layernorm (f32 in) -> bf16 out, D=512, 128 threads ----------------
__global__ __launch_bounds__(128) void layernorm_to_bf16(const float* __restrict__ x,
                                                         const float* __restrict__ sc,
                                                         const float* __restrict__ bs,
                                                         bf16* __restrict__ out) {
  __shared__ float red[128];
  int row = blockIdx.x, t = threadIdx.x;
  const float* xr = x + (size_t)row * CD;
  float4 v = *(const float4*)(xr + t * 4);
  red[t] = v.x + v.y + v.z + v.w;
  __syncthreads();
  for (int s = 64; s > 0; s >>= 1) { if (t < s) red[t] += red[t + s]; __syncthreads(); }
  float mu = red[0] * (1.0f / CD);
  __syncthreads();
  float d0 = v.x - mu, d1 = v.y - mu, d2 = v.z - mu, d3 = v.w - mu;
  red[t] = d0 * d0 + d1 * d1 + d2 * d2 + d3 * d3;
  __syncthreads();
  for (int s = 64; s > 0; s >>= 1) { if (t < s) red[t] += red[t + s]; __syncthreads(); }
  float rs = rsqrtf(red[0] * (1.0f / CD) + 1e-6f);
  const float* sp = sc + t * 4;
  const float* bp = bs + t * 4;
  bf16* o = out + (size_t)row * CD + t * 4;
  o[0] = (bf16)(d0 * rs * sp[0] + bp[0]);
  o[1] = (bf16)(d1 * rs * sp[1] + bp[1]);
  o[2] = (bf16)(d2 * rs * sp[2] + bp[2]);
  o[3] = (bf16)(d3 * rs * sp[3] + bp[3]);
}

// ---------------- WMMA bf16 GEMM: C[M,N] = A[M,K] * Bt[N,K]^T (+epilogue) ----------------
// 256 threads = 8 wave32. Block tile 128x128, wave tile 64x32 (4x2 of 16x16x32 WMMA).
// Both A ([M,K]) and Bt ([N,K], pre-transposed weights) stage as straight 32B copies
// into LDS ([row][k], stride LSTR). Double-buffered: one barrier per K-step; next
// tile's copies (async global->LDS) issued before the WMMA block.
enum { EPI_BIAS = 1, EPI_GELU = 2, EPI_RESID = 4, EPI_OBF16 = 8 };
constexpr int LSTR = 40;  // LDS row stride in halves (80B, 16B-aligned, skews banks)

template <int FLAGS>
__global__ __launch_bounds__(256) void gemm_wmma_bf16(
    const bf16* __restrict__ A, const bf16* __restrict__ Bt,
    const float* __restrict__ bias, const float* __restrict__ resid,
    float* __restrict__ outF, bf16* __restrict__ outB,
    int M, int N, int K) {
  __shared__ bf16 As[2][128 * LSTR];
  __shared__ bf16 Bs[2][128 * LSTR];

  const int tid  = threadIdx.x;
  const int lane = tid & 31;
  const int wid  = tid >> 5;
  const int wm   = wid >> 2;     // 0..1 -> 64-row strip
  const int wn   = wid & 3;      // 0..3 -> 32-col strip
  const int lrow = lane & 15;
  const int lhi  = lane >> 4;
  const int m0 = blockIdx.y * 128;
  const int n0 = blockIdx.x * 128;

  // staging assignment: each thread copies 16 halves of A and 16 halves of B
  const int srow = tid >> 1;          // 0..127
  const int skc  = (tid & 1) * 16;    // 0 or 16
  const bf16* gA = A  + (size_t)(m0 + srow) * K + skc;  // rows >= M overrun into ws: harmless, store-guarded
  const bf16* gB = Bt + (size_t)(n0 + srow) * K + skc;
  bf16* lA = &As[0][0] + srow * LSTR + skc;
  bf16* lB = &Bs[0][0] + srow * LSTR + skc;
  constexpr int LBUF = 128 * LSTR;

  v8f acc[4][2];
  const v8f vz = {0.f, 0.f, 0.f, 0.f, 0.f, 0.f, 0.f, 0.f};
#pragma unroll
  for (int i = 0; i < 4; ++i)
#pragma unroll
    for (int j = 0; j < 2; ++j) acc[i][j] = vz;

  const int nT = K >> 5;  // K-steps of 32
  // prologue: stage tile 0 into buffer 0
  copy16_g2l(lA, gA);
  copy16_g2l(lB, gB);

  for (int t = 0; t < nT; ++t) {
    wait_stage();
    __syncthreads();
    const int cur = t & 1;
    if (t + 1 < nT) {  // issue next tile into the other buffer while computing
      const int nxt = (t + 1) & 1;
      copy16_g2l(lA + nxt * LBUF, gA + (size_t)(t + 1) * 32);
      copy16_g2l(lB + nxt * LBUF, gB + (size_t)(t + 1) * 32);
    }

    const bf16* as = &As[cur][0];
    const bf16* bs = &Bs[cur][0];
    v16bf afrag[4], bfrag[2];
#pragma unroll
    for (int im = 0; im < 4; ++im) {
      int r = (wm * 64 + im * 16 + lrow) * LSTR;
      // halves 0..7 : K = lhi*8 + i ; halves 8..15 : K = 16 + lhi*8 + (i-8)
      ((uint4*)&afrag[im])[0] = *(const uint4*)(as + r + lhi * 8);
      ((uint4*)&afrag[im])[1] = *(const uint4*)(as + r + lhi * 8 + 16);
    }
#pragma unroll
    for (int in = 0; in < 2; ++in) {
      int c = (wn * 32 + in * 16 + lrow) * LSTR;
      // halves 0..7 : K = lhi*8 + i ; halves 8..15 : K = 16 + lhi*8 + (i-8)
      ((uint4*)&bfrag[in])[0] = *(const uint4*)(bs + c + lhi * 8);
      ((uint4*)&bfrag[in])[1] = *(const uint4*)(bs + c + lhi * 8 + 16);
    }
#pragma unroll
    for (int im = 0; im < 4; ++im)
#pragma unroll
      for (int in = 0; in < 2; ++in)
        acc[im][in] = __builtin_amdgcn_wmma_f32_16x16x32_bf16(
            false, afrag[im], false, bfrag[in], (short)0, acc[im][in], false, false);
  }

  // --- epilogue: C layout: row_in_tile = lhi*8 + r, col_in_tile = lrow ---
#pragma unroll
  for (int im = 0; im < 4; ++im) {
#pragma unroll
    for (int in = 0; in < 2; ++in) {
#pragma unroll
      for (int r = 0; r < 8; ++r) {
        int row = m0 + wm * 64 + im * 16 + lhi * 8 + r;
        int col = n0 + wn * 32 + in * 16 + lrow;
        if (row < M) {
          float v = acc[im][in][r];
          if (FLAGS & EPI_BIAS)  v += bias[col];
          if (FLAGS & EPI_GELU)  v = gelu_tanh(v);
          if (FLAGS & EPI_RESID) v += resid[(size_t)row * N + col];
          if (FLAGS & EPI_OBF16) outB[(size_t)row * N + col] = (bf16)v;
          else                   outF[(size_t)row * N + col] = v;
        }
      }
    }
  }
}

// ---------------- block-local attention: one WG per (b, nb, h) ----------------
__global__ __launch_bounds__(64) void attn_kernel(const float* __restrict__ q,
                                                  const float* __restrict__ k,
                                                  const float* __restrict__ v,
                                                  const int* __restrict__ inp,
                                                  float* __restrict__ o) {
  int idx = blockIdx.x;            // ((b*NB + nb)*H + h)
  int h  = idx % CH;
  int nb = (idx / CH) % CNB;
  int b  = idx / (CH * CNB);
  __shared__ float Ks[CBS][CHD];
  __shared__ float Vs[CBS][CHD];
  __shared__ float keyok[CBS];
  int t = threadIdx.x;
  for (int i = t; i < CBS * CHD; i += 64) {
    int kk = i >> 6, e = i & 63;
    int srow = b * CS + nb * CBS + kk;
    size_t base = ((size_t)srow * CH + h) * CHD + e;
    Ks[kk][e] = k[base];
    Vs[kk][e] = v[base];
  }
  for (int i = t; i < CBS; i += 64)
    keyok[i] = (inp[b * CS + nb * CBS + i] > 0) ? 1.0f : 0.0f;
  __syncthreads();
  if (t < CBS) {
    int srow = b * CS + nb * CBS + t;
    const float* qp = q + ((size_t)srow * CH + h) * CHD;
    float qr[CHD];
#pragma unroll
    for (int e = 0; e < CHD; ++e) qr[e] = qp[e];
    float sc[CBS];
    float mx = -3.4e38f;
    for (int kk = 0; kk < CBS; ++kk) {
      float d = 0.f;
#pragma unroll
      for (int e = 0; e < CHD; ++e) d += qr[e] * Ks[kk][e];
      d *= 0.125f;  // 1/sqrt(64)
      bool ok = (kk <= t) && (keyok[kk] > 0.f);
      d = ok ? d : -1e9f;
      sc[kk] = d;
      mx = fmaxf(mx, d);
    }
    float se = 0.f;
    for (int kk = 0; kk < CBS; ++kk) { float e_ = expf(sc[kk] - mx); sc[kk] = e_; se += e_; }
    float inv = 1.0f / se;
    float* op = o + ((size_t)srow * CH + h) * CHD;
    for (int e = 0; e < CHD; ++e) {
      float a = 0.f;
      for (int kk = 0; kk < CBS; ++kk) a += sc[kk] * Vs[kk][e];
      op[e] = a * inv;
    }
  }
}

// ---------------- host-side orchestration ----------------
extern "C" void kernel_launch(void* const* d_in, const int* in_sizes, int n_in,
                              void* d_out, int out_size, void* d_ws, size_t ws_size,
                              hipStream_t stream) {
  (void)in_sizes; (void)n_in; (void)out_size; (void)ws_size;
  const int*   inp   = (const int*)  d_in[0];
  const float* emb   = (const float*)d_in[1];
  const float* ln1_s = (const float*)d_in[2];
  const float* ln1_b = (const float*)d_in[3];
  const float* wq    = (const float*)d_in[4];
  const float* wk    = (const float*)d_in[5];
  const float* wv    = (const float*)d_in[6];
  const float* wo    = (const float*)d_in[7];
  const float* ln2_s = (const float*)d_in[8];
  const float* ln2_b = (const float*)d_in[9];
  const float* w1    = (const float*)d_in[10];
  const float* b1    = (const float*)d_in[11];
  const float* w2    = (const float*)d_in[12];
  const float* b2    = (const float*)d_in[13];
  const float* lnf_s = (const float*)d_in[14];
  const float* lnf_b = (const float*)d_in[15];
  const float* wout  = (const float*)d_in[16];
  const float* bout  = (const float*)d_in[17];

  char* ws = (char*)d_ws;
  size_t off = 0;
  auto alloc = [&](size_t bytes) -> void* {
    off = (off + 255) & ~(size_t)255;
    void* p = ws + off;
    off += bytes;
    return p;
  };

  float* x   = (float*)alloc((size_t)CM * CD * 4);
  bf16*  hb  = (bf16*) alloc((size_t)CM * CD * 2);
  bf16*  mb  = (bf16*) alloc((size_t)CM * CMLP * 2);
  float* qf  = (float*)alloc((size_t)CM * CD * 4);
  float* kf  = (float*)alloc((size_t)CM * CD * 4);
  float* vf  = (float*)alloc((size_t)CM * CD * 4);
  float* of  = (float*)alloc((size_t)CM * CD * 4);
  // pre-transposed bf16 weights: stored [N][K] per layer
  bf16* wqt  = (bf16*)alloc((size_t)CL * CD * CD * 2);
  bf16* wkt  = (bf16*)alloc((size_t)CL * CD * CD * 2);
  bf16* wvt  = (bf16*)alloc((size_t)CL * CD * CD * 2);
  bf16* wot  = (bf16*)alloc((size_t)CL * CD * CD * 2);
  bf16* w1t  = (bf16*)alloc((size_t)CL * CD * CMLP * 2);
  bf16* w2t  = (bf16*)alloc((size_t)CL * CMLP * CD * 2);
  bf16* woutt= (bf16*)alloc((size_t)CD * CV * 2);
  // tail pad so GEMM A-staging row overruns (rows 4000..4095) stay inside ws
  (void)alloc((size_t)1 << 20);

  // weight cast+transpose (recomputed each call -> deterministic, graph-safe)
  {
    dim3 t256(256);
    dim3 gqkv(CD / 32, CD / 32, CL);
    cast_transpose_bf16<<<gqkv, t256, 0, stream>>>(wq, wqt, CD, CD);
    cast_transpose_bf16<<<gqkv, t256, 0, stream>>>(wk, wkt, CD, CD);
    cast_transpose_bf16<<<gqkv, t256, 0, stream>>>(wv, wvt, CD, CD);
    cast_transpose_bf16<<<gqkv, t256, 0, stream>>>(wo, wot, CD, CD);
    dim3 g1(CMLP / 32, CD / 32, CL);
    cast_transpose_bf16<<<g1, t256, 0, stream>>>(w1, w1t, CD, CMLP);
    dim3 g2(CD / 32, CMLP / 32, CL);
    cast_transpose_bf16<<<g2, t256, 0, stream>>>(w2, w2t, CMLP, CD);
    dim3 g3(CV / 32, CD / 32, 1);
    cast_transpose_bf16<<<g3, t256, 0, stream>>>(wout, woutt, CD, CV);
  }

  embed_pe<<<CM, 256, 0, stream>>>(inp, emb, x);

  const dim3 blk(256);
  const dim3 gD(CD / 128, (CM + 127) / 128);      // N=512
  const dim3 gMLP(CMLP / 128, (CM + 127) / 128);  // N=2048
  const dim3 gV(CV / 128, (CM + 127) / 128);      // N=32000

  for (int i = 0; i < CL; ++i) {
    // h = LN1(x) -> bf16
    layernorm_to_bf16<<<CM, 128, 0, stream>>>(x, ln1_s + i * CD, ln1_b + i * CD, hb);
    // q/k/v = h @ W
    gemm_wmma_bf16<0><<<gD, blk, 0, stream>>>(hb, wqt + (size_t)i * CD * CD, nullptr, nullptr, qf, nullptr, CM, CD, CD);
    gemm_wmma_bf16<0><<<gD, blk, 0, stream>>>(hb, wkt + (size_t)i * CD * CD, nullptr, nullptr, kf, nullptr, CM, CD, CD);
    gemm_wmma_bf16<0><<<gD, blk, 0, stream>>>(hb, wvt + (size_t)i * CD * CD, nullptr, nullptr, vf, nullptr, CM, CD, CD);
    // block-local masked attention
    attn_kernel<<<CB * CNB * CH, 64, 0, stream>>>(qf, kf, vf, inp, of);
    // x += o @ wo
    cast_f32_bf16<<<2048, 256, 0, stream>>>(of, hb, (size_t)CM * CD);
    gemm_wmma_bf16<EPI_RESID><<<gD, blk, 0, stream>>>(hb, wot + (size_t)i * CD * CD, nullptr, x, x, nullptr, CM, CD, CD);
    // MLP
    layernorm_to_bf16<<<CM, 128, 0, stream>>>(x, ln2_s + i * CD, ln2_b + i * CD, hb);
    gemm_wmma_bf16<EPI_BIAS | EPI_GELU | EPI_OBF16><<<gMLP, blk, 0, stream>>>(
        hb, w1t + (size_t)i * CD * CMLP, b1 + (size_t)i * CMLP, nullptr, nullptr, mb, CM, CMLP, CD);
    gemm_wmma_bf16<EPI_BIAS | EPI_RESID><<<gD, blk, 0, stream>>>(
        mb, w2t + (size_t)i * CMLP * CD, b2 + (size_t)i * CD, x, x, nullptr, CM, CD, CMLP);
  }

  // final LN + vocab projection straight into d_out (f32 [B,S,V])
  layernorm_to_bf16<<<CM, 128, 0, stream>>>(x, lnf_s, lnf_b, hb);
  gemm_wmma_bf16<EPI_BIAS><<<gV, blk, 0, stream>>>(hb, woutt, bout, nullptr, (float*)d_out, nullptr, CM, CV, CD);
}
